// SPSM_37297495998698
// MI455X (gfx1250) — matrix operand order repaired and checked by reading, not compiled, not run on hardware
//
#include <hip/hip_runtime.h>
#include <hip/hip_bf16.h>

// MI455X / gfx1250, wave32. Contractions on v_wmma_f32_16x16x32_f16.
// B-side operands pre-packed to fragment layout (2x b128/frag); A tiles staged to LDS
// via CDNA5 async-to-LDS DMA (global_load_async_to_lds_b128 + s_wait_asynccnt).

typedef _Float16 half_t;
typedef __attribute__((ext_vector_type(16))) _Float16     v16h;
typedef __attribute__((ext_vector_type(8)))  float        v8f;
typedef __attribute__((ext_vector_type(8)))  unsigned int v8u;
typedef __attribute__((address_space(3)))    half_t*      lds_ptr_t;

#define B_  8
#define P_  1000
#define N_  1000
#define E_  128
#define H_  8
#define D_  16
#define HD_ 128
#define NT_ 63          // ceil(1000/16)
#define KQ_ 160         // 132 padded to 5 chunks of 32

// A-fragment K offset (16-bit A 16x32, contiguous order matching packed B): VGPR j -> 2j+16h
__device__ __forceinline__ int koffA(int j, int h) {
  return 2 * j + 16 * h;
}

__device__ __forceinline__ v8f wmma_f16(v16h a, v16h b, v8f c) {
  return __builtin_amdgcn_wmma_f32_16x16x32_f16(false, a, false, b, (short)0, c, false, false);
}

// packed fragment: 16 halves contiguous per lane (32B) -> 2x b128
__device__ __forceinline__ v16h load_frag32(const half_t* __restrict__ p) {
  v8u t = *reinterpret_cast<const v8u*>(p);
  return __builtin_bit_cast(v16h, t);
}
// A fragment from f16 row (pairs ko,ko+1 adjacent; 16B runs merge to b128)
__device__ __forceinline__ v16h load_arow(const half_t* base, int kb, int hh) {
  v8u u;
#pragma unroll
  for (int j = 0; j < 8; ++j)
    u[j] = *reinterpret_cast<const unsigned int*>(base + kb + koffA(j, hh));
  return __builtin_bit_cast(v16h, u);
}

// CDNA5 async DMA: one 16B global->LDS transfer per lane (ASYNCcnt-tracked)
__device__ __forceinline__ void async_g2l_b128(half_t* lds_dst, const half_t* gsrc) {
  unsigned lofs = (unsigned)(unsigned long long)(lds_ptr_t)lds_dst;
  unsigned long long g = (unsigned long long)gsrc;
  asm volatile("global_load_async_to_lds_b128 %0, %1, off" :: "v"(lofs), "v"(g) : "memory");
}
__device__ __forceinline__ void wait_async0() {
  asm volatile("s_wait_asynccnt 0x0" ::: "memory");
}

__device__ __forceinline__ float redmax16(float x) {
  x = fmaxf(x, __shfl_xor(x, 1, 16));
  x = fmaxf(x, __shfl_xor(x, 2, 16));
  x = fmaxf(x, __shfl_xor(x, 4, 16));
  x = fmaxf(x, __shfl_xor(x, 8, 16));
  return x;
}
__device__ __forceinline__ float redsum16(float x) {
  x += __shfl_xor(x, 1, 16);
  x += __shfl_xor(x, 2, 16);
  x += __shfl_xor(x, 4, 16);
  x += __shfl_xor(x, 8, 16);
  return x;
}

// ---------------------------------------------------------------- small prep kernels
__global__ void k_cvt(const float* __restrict__ src, half_t* __restrict__ dst, int n) {
  int i = blockIdx.x * blockDim.x + threadIdx.x;
  if (i < n) dst[i] = (half_t)src[i];
}

__global__ void k_prep_q(const float* __restrict__ eln, const float* __restrict__ attr,
                         half_t* __restrict__ qin) {
  int t = blockIdx.x * blockDim.x + threadIdx.x;
  if (t >= B_ * P_ * KQ_) return;
  int row = t / KQ_, k = t % KQ_;
  float v = (k < E_) ? eln[(long)row * E_ + k]
          : (k < E_ + 4) ? attr[(long)row * 4 + (k - E_)] : 0.f;
  qin[t] = (half_t)v;
}

__global__ void k_pack_w(const float* __restrict__ W, int Kvalid, int KC,
                         half_t* __restrict__ out) {
  int t = blockIdx.x * blockDim.x + threadIdx.x;
  if (t >= 8 * KC * 512) return;
  int e = t & 15, lane = (t >> 4) & 31;
  int rest = t >> 9, kc = rest % KC, ct = rest / KC;
  int kr = kc * 32 + 2 * (e >> 1) + 16 * (lane >> 4) + (e & 1);
  int col = ct * 16 + (lane & 15);
  out[t] = (kr < Kvalid) ? (half_t)W[(long)kr * HD_ + col] : (half_t)0.f;
}

__global__ void k_pack_e(const float* __restrict__ enc, half_t* __restrict__ out) {
  int t = blockIdx.x * blockDim.x + threadIdx.x;
  if (t >= B_ * NT_ * 4 * 512) return;
  int e = t & 15, lane = (t >> 4) & 31, kc = (t >> 9) & 3;
  int rest = t >> 11, nt = rest % NT_, b = rest / NT_;
  int kr = kc * 32 + 2 * (e >> 1) + 16 * (lane >> 4) + (e & 1);
  int n = min(nt * 16 + (lane & 15), N_ - 1);
  out[t] = (half_t)enc[((long)b * N_ + n) * E_ + kr];
}

// ---------------------------------------------------------------- KV projections
// A tile (16x128 f16, contiguous 4KB) staged via async DMA; 4 outputs written directly
// into attention's packed fragment layouts (padding lanes pre-zeroed by memset).
__global__ void k_proj_kv(const half_t* __restrict__ enc16,
                          const half_t* __restrict__ pWks, const half_t* __restrict__ pWvs,
                          const half_t* __restrict__ pWkt, const half_t* __restrict__ pWvt,
                          half_t* __restrict__ pks, half_t* __restrict__ pvs,
                          half_t* __restrict__ pkt, half_t* __restrict__ pvt) {
  __shared__ __align__(16) half_t lds_a[16 * E_];
  const int t = threadIdx.x;
  const int wave = t >> 5, lane = t & 31;
  const int nl = lane & 15, hh = lane >> 4;
  const int b = blockIdx.x / NT_, ntile = blockIdx.x % NT_;
  const half_t* tile = enc16 + ((long)b * N_ + ntile * 16) * E_;
  async_g2l_b128(&lds_a[t * 8], tile + t * 8);
  wait_async0();
  __syncthreads();

  v8f c0 = {}, c1 = {}, c2 = {}, c3 = {};
#pragma unroll
  for (int kc = 0; kc < 4; ++kc) {
    v16h a = load_arow(lds_a + nl * E_, kc * 32, hh);
    const long wo = ((long)(wave * 4 + kc) * 32 + lane) * 16;
    c0 = wmma_f16(a, load_frag32(pWks + wo), c0);
    c1 = wmma_f16(a, load_frag32(pWvs + wo), c1);
    c2 = wmma_f16(a, load_frag32(pWkt + wo), c2);
    c3 = wmma_f16(a, load_frag32(pWvt + wo), c3);
  }
  const long tbase = ((long)(b * H_ + wave) * NT_ + ntile) * 512;
#pragma unroll
  for (int r = 0; r < 8; ++r) {
    int row = ntile * 16 + r + 8 * hh;
    if (row < N_) {
      int e = row & 15;
      pks[tbase + e * 16 + nl] = (half_t)c0[r];   // k-frag: lane = n&15, elem = d
      pkt[tbase + e * 16 + nl] = (half_t)c2[r];
      pvs[tbase + nl * 16 + e] = (half_t)c1[r];   // v-frag: lane = d,    elem = n&15
      pvt[tbase + nl * 16 + e] = (half_t)c3[r];
    }
  }
}

// ---------------------------------------------------------------- Q / gate projections
__global__ void k_proj_q(const half_t* __restrict__ qin,
                         const half_t* __restrict__ pWqs, const half_t* __restrict__ pWqt,
                         const half_t* __restrict__ pWg, const float* __restrict__ bg,
                         half_t* __restrict__ qs, half_t* __restrict__ qt,
                         float* __restrict__ gate) {
  __shared__ __align__(16) half_t lds_a[16 * KQ_];
  const int t = threadIdx.x;
  const int wave = t >> 5, lane = t & 31;
  const int nl = lane & 15, hh = lane >> 4;
  const half_t* tile = qin + (long)blockIdx.x * 16 * KQ_;
  async_g2l_b128(&lds_a[t * 8], tile + t * 8);
  if (t < 64) async_g2l_b128(&lds_a[2048 + t * 8], tile + 2048 + t * 8);
  wait_async0();
  __syncthreads();

  const int nb = wave * 16 + nl;
  v8f c0 = {}, c1 = {}, c2 = {};
#pragma unroll
  for (int kc = 0; kc < 5; ++kc) {
    v16h a = load_arow(lds_a + nl * KQ_, kc * 32, hh);
    const long wo = ((long)(wave * 5 + kc) * 32 + lane) * 16;
    c0 = wmma_f16(a, load_frag32(pWqs + wo), c0);
    c1 = wmma_f16(a, load_frag32(pWqt + wo), c1);
    c2 = wmma_f16(a, load_frag32(pWg + wo), c2);
  }
#pragma unroll
  for (int r = 0; r < 8; ++r) {
    long idx = ((long)blockIdx.x * 16 + r + 8 * hh) * HD_ + nb;
    qs[idx] = (half_t)c0[r];
    qt[idx] = (half_t)c1[r];
    float x = c2[r] + bg[nb];
    gate[idx] = 1.f / (1.f + __expf(-x));
  }
}

// ---------------------------------------------------------------- fused dual flash attention
// Both (shared,task) attentions per wave: one ninf_mask stream serves both (halves the
// dominant 64MB mask traffic). LDS prob tile reused sequentially (same-wave DS in-order).
__global__ void k_attn2(const half_t* __restrict__ qs16, const half_t* __restrict__ qt16,
                        const half_t* __restrict__ pks, const half_t* __restrict__ pvs,
                        const half_t* __restrict__ pkt, const half_t* __restrict__ pvt,
                        const float* __restrict__ mask,
                        half_t* __restrict__ os16, half_t* __restrict__ ot16) {
  __shared__ half_t lds_w[8][16][18];
  const int wave = threadIdx.x >> 5, lane = threadIdx.x & 31;
  const int nl = lane & 15, hh = lane >> 4;
  const int g = blockIdx.x * 8 + wave;
  const int b = g / (H_ * NT_);
  const int rem = g % (H_ * NT_);
  const int h = rem / NT_;
  const int pt = rem % NT_;

  const int  pA = min(pt * 16 + nl, P_ - 1);
  const long qoff = ((long)b * P_ + pA) * HD_ + h * D_;
  v8u uqs = {}, uqt = {};
#pragma unroll
  for (int j = 0; j < 8; ++j) {
    int ko = koffA(j, hh);
    if (ko < D_) {
      uqs[j] = *reinterpret_cast<const unsigned int*>(qs16 + qoff + ko);
      uqt[j] = *reinterpret_cast<const unsigned int*>(qt16 + qoff + ko);
    }
  }
  const v16h aqs = __builtin_bit_cast(v16h, uqs);
  const v16h aqt = __builtin_bit_cast(v16h, uqt);

  const long fo = ((long)(b * H_ + h) * NT_) * 512 + lane * 16;
  const half_t* ks = pks + fo;  const half_t* vs = pvs + fo;
  const half_t* kt = pkt + fo;  const half_t* vt = pvt + fo;

  v8f os = {}, ot = {};
  float ms[8], ls[8], mt[8], lt[8];
#pragma unroll
  for (int r = 0; r < 8; ++r) {
    ms[r] = -__builtin_inff(); ls[r] = 0.f;
    mt[r] = -__builtin_inff(); lt[r] = 0.f;
  }

  for (int nt = 0; nt < NT_; ++nt) {
    if (nt + 1 < NT_) {
      __builtin_prefetch(ks + (nt + 1) * 512, 0, 1);
      __builtin_prefetch(vs + (nt + 1) * 512, 0, 1);
      __builtin_prefetch(kt + (nt + 1) * 512, 0, 1);
      __builtin_prefetch(vt + (nt + 1) * 512, 0, 1);
    }
    v8f ss = {}, st = {};
    ss = wmma_f16(aqs, load_frag32(ks + nt * 512), ss);
    st = wmma_f16(aqt, load_frag32(kt + nt * 512), st);
    const int  ng = nt * 16 + nl;
    const bool nv = ng < N_;
    const int  ngc = min(ng, N_ - 1);
    float mv[8];
#pragma unroll
    for (int r = 0; r < 8; ++r) {
      int pc = min(pt * 16 + r + 8 * hh, P_ - 1);
      mv[r] = mask[((long)b * P_ + pc) * (long)N_ + ngc];   // single mask stream
    }
    // ---- shared head
#pragma unroll
    for (int r = 0; r < 8; ++r) {
      float sv = nv ? (ss[r] * 0.25f + mv[r]) : -__builtin_inff();
      float mn = fmaxf(ms[r], redmax16(sv));
      float sc = __expf(ms[r] - mn);
      float w  = __expf(sv - mn);
      ls[r] = ls[r] * sc + redsum16(w);
      ms[r] = mn;
      os[r] *= sc;
      lds_w[wave][r + 8 * hh][nl] = (half_t)w;
    }
    asm volatile("" ::: "memory");
    __builtin_amdgcn_wave_barrier();
    {
      v8u uw = {};
#pragma unroll
      for (int j = 0; j < 8; ++j) {
        int ko = koffA(j, hh);
        if (ko < 16) uw[j] = *reinterpret_cast<const unsigned int*>(&lds_w[wave][nl][ko]);
      }
      os = wmma_f16(__builtin_bit_cast(v16h, uw), load_frag32(vs + nt * 512), os);
    }
    // ---- task head (reuse LDS tile; same-wave DS ops are in-order)
    asm volatile("" ::: "memory");
    __builtin_amdgcn_wave_barrier();
#pragma unroll
    for (int r = 0; r < 8; ++r) {
      float sv = nv ? (st[r] * 0.25f + mv[r]) : -__builtin_inff();
      float mn = fmaxf(mt[r], redmax16(sv));
      float sc = __expf(mt[r] - mn);
      float w  = __expf(sv - mn);
      lt[r] = lt[r] * sc + redsum16(w);
      mt[r] = mn;
      ot[r] *= sc;
      lds_w[wave][r + 8 * hh][nl] = (half_t)w;
    }
    asm volatile("" ::: "memory");
    __builtin_amdgcn_wave_barrier();
    {
      v8u uw = {};
#pragma unroll
      for (int j = 0; j < 8; ++j) {
        int ko = koffA(j, hh);
        if (ko < 16) uw[j] = *reinterpret_cast<const unsigned int*>(&lds_w[wave][nl][ko]);
      }
      ot = wmma_f16(__builtin_bit_cast(v16h, uw), load_frag32(vt + nt * 512), ot);
    }
  }
#pragma unroll
  for (int r = 0; r < 8; ++r) {
    int p = pt * 16 + r + 8 * hh;
    if (p < P_) {
      long oi = ((long)b * P_ + p) * HD_ + h * D_ + nl;
      os16[oi] = (half_t)(os[r] / ls[r]);
      ot16[oi] = (half_t)(ot[r] / lt[r]);
    }
  }
}

// ---------------------------------------------------------------- output proj + gated combine
__global__ void k_combine(const half_t* __restrict__ as16, const half_t* __restrict__ at16,
                          const half_t* __restrict__ pWcs, const float* __restrict__ bcs,
                          const half_t* __restrict__ pWct, const float* __restrict__ bct,
                          const float* __restrict__ gate, half_t* __restrict__ out16) {
  __shared__ __align__(16) half_t lds_st[2 * 16 * HD_];
  const int t = threadIdx.x;
  const int wave = t >> 5, lane = t & 31;
  const int nl = lane & 15, hh = lane >> 4;
  async_g2l_b128(&lds_st[t * 8],        as16 + (long)blockIdx.x * 16 * HD_ + t * 8);
  async_g2l_b128(&lds_st[2048 + t * 8], at16 + (long)blockIdx.x * 16 * HD_ + t * 8);
  wait_async0();
  __syncthreads();

  const int nb = wave * 16 + nl;
  v8f cs = {}, ct = {};
#pragma unroll
  for (int kc = 0; kc < 4; ++kc) {
    const long wo = ((long)(wave * 4 + kc) * 32 + lane) * 16;
    cs = wmma_f16(load_arow(lds_st + nl * HD_, kc * 32, hh),        load_frag32(pWcs + wo), cs);
    ct = wmma_f16(load_arow(lds_st + 2048 + nl * HD_, kc * 32, hh), load_frag32(pWct + wo), ct);
  }
#pragma unroll
  for (int r = 0; r < 8; ++r) {
    long idx = ((long)blockIdx.x * 16 + r + 8 * hh) * HD_ + nb;
    float gv = gate[idx];
    out16[idx] = (half_t)(gv * (cs[r] + bcs[nb]) + (1.f - gv) * (ct[r] + bct[nb]));
  }
}

// ---------------------------------------------------------------- pointer: scores+clip+softmax
__global__ void k_pointer(const half_t* __restrict__ g16, const half_t* __restrict__ pe,
                          const float* __restrict__ mask, float* __restrict__ out) {
  const int wave = threadIdx.x >> 5, lane = threadIdx.x & 31;
  const int nl = lane & 15, hh = lane >> 4;
  const int gidx = blockIdx.x * 8 + wave;
  const int b = gidx / NT_;
  const int pt = gidx % NT_;

  const int  pA = min(pt * 16 + nl, P_ - 1);
  const half_t* gbase = g16 + ((long)b * P_ + pA) * HD_;
  v16h ag[4];
#pragma unroll
  for (int kc = 0; kc < 4; ++kc) ag[kc] = load_arow(gbase, kc * 32, hh);

  const half_t* etile = pe + (long)b * NT_ * 4 * 512 + lane * 16;

  float mr[8], lr[8];
#pragma unroll
  for (int r = 0; r < 8; ++r) { mr[r] = -__builtin_inff(); lr[r] = 0.f; }

  for (int nt = 0; nt < NT_; ++nt) {
    if (nt + 1 < NT_) __builtin_prefetch(etile + (nt + 1) * 2048, 0, 1);
    v8f s = {};
#pragma unroll
    for (int kc = 0; kc < 4; ++kc)
      s = wmma_f16(ag[kc], load_frag32(etile + nt * 2048 + kc * 512), s);
    const int  ng = nt * 16 + nl;
    const bool nv = ng < N_;
    const int  ngc = min(ng, N_ - 1);
#pragma unroll
    for (int r = 0; r < 8; ++r) {
      int pc = min(pt * 16 + r + 8 * hh, P_ - 1);
      float sv = nv ? (10.f * tanhf(s[r] * 0.08838834764831845f)
                       + mask[((long)b * P_ + pc) * (long)N_ + ngc])
                    : -__builtin_inff();
      float mn = fmaxf(mr[r], redmax16(sv));
      float sc = __expf(mr[r] - mn);
      float w  = __expf(sv - mn);
      lr[r] = lr[r] * sc + redsum16(w);
      mr[r] = mn;
    }
  }
  float invl[8];
#pragma unroll
  for (int r = 0; r < 8; ++r) invl[r] = 1.f / lr[r];

  for (int nt = 0; nt < NT_; ++nt) {          // pass 2: recompute + write probs
    v8f s = {};
#pragma unroll
    for (int kc = 0; kc < 4; ++kc)
      s = wmma_f16(ag[kc], load_frag32(etile + nt * 2048 + kc * 512), s);
    const int  ng = nt * 16 + nl;
    const bool nv = ng < N_;
    const int  ngc = min(ng, N_ - 1);
#pragma unroll
    for (int r = 0; r < 8; ++r) {
      int p = pt * 16 + r + 8 * hh;
      if (p < P_ && nv) {
        float sv = 10.f * tanhf(s[r] * 0.08838834764831845f)
                 + mask[((long)b * P_ + p) * (long)N_ + ngc];
        out[((long)b * P_ + p) * (long)N_ + ng] = __expf(sv - mr[r]) * invl[r];
      }
    }
  }
}

// ---------------------------------------------------------------- launch
extern "C" void kernel_launch(void* const* d_in, const int* in_sizes, int n_in,
                              void* d_out, int out_size, void* d_ws, size_t ws_size,
                              hipStream_t stream) {
  const float* eln  = (const float*)d_in[0];
  const float* attr = (const float*)d_in[1];
  const float* enc  = (const float*)d_in[2];
  const float* mask = (const float*)d_in[3];
  const float* Wqs  = (const float*)d_in[4];
  const float* Wks  = (const float*)d_in[5];
  const float* Wvs  = (const float*)d_in[6];
  const float* Wqt  = (const float*)d_in[7];
  const float* Wkt  = (const float*)d_in[8];
  const float* Wvt  = (const float*)d_in[9];
  const float* Wcs  = (const float*)d_in[10];
  const float* bcs  = (const float*)d_in[11];
  const float* Wct  = (const float*)d_in[12];
  const float* bct  = (const float*)d_in[13];
  const float* Wg   = (const float*)d_in[14];
  const float* bg   = (const float*)d_in[15];

  char* ws = (char*)d_ws;
  const size_t LBUF = (size_t)B_ * N_ * HD_ * sizeof(half_t);
  const size_t PKV  = (size_t)B_ * H_ * NT_ * 512 * sizeof(half_t);
  const size_t PE   = (size_t)B_ * NT_ * 4 * 512 * sizeof(half_t);
  const size_t PW4  = (size_t)8 * 4 * 512 * sizeof(half_t);
  const size_t PW5  = (size_t)8 * 5 * 512 * sizeof(half_t);

  half_t* enc16 = (half_t*)ws; ws += LBUF;
  half_t* qin   = (half_t*)ws; ws += (size_t)B_ * P_ * KQ_ * sizeof(half_t);
  half_t* pks   = (half_t*)ws; ws += PKV;      // 4 packed KV buffers contiguous
  half_t* pvs   = (half_t*)ws; ws += PKV;
  half_t* pkt   = (half_t*)ws; ws += PKV;
  half_t* pvt   = (half_t*)ws; ws += PKV;
  half_t* pe    = (half_t*)ws; ws += PE;
  half_t* qs    = (half_t*)ws; ws += LBUF;
  half_t* qt    = (half_t*)ws; ws += LBUF;
  half_t* aso   = (half_t*)ws; ws += LBUF;
  half_t* ato   = (half_t*)ws; ws += LBUF;
  half_t* g16   = (half_t*)ws; ws += LBUF;
  float*  gate  = (float*)ws;  ws += (size_t)B_ * P_ * HD_ * sizeof(float);
  half_t* pWks  = (half_t*)ws; ws += PW4;
  half_t* pWvs  = (half_t*)ws; ws += PW4;
  half_t* pWkt  = (half_t*)ws; ws += PW4;
  half_t* pWvt  = (half_t*)ws; ws += PW4;
  half_t* pWcs  = (half_t*)ws; ws += PW4;
  half_t* pWct  = (half_t*)ws; ws += PW4;
  half_t* pWqs  = (half_t*)ws; ws += PW5;
  half_t* pWqt  = (half_t*)ws; ws += PW5;
  half_t* pWg   = (half_t*)ws; ws += PW5;

  const int NE = B_ * N_ * HD_;
  k_cvt<<<(NE + 255) / 256, 256, 0, stream>>>(enc, enc16, NE);
  k_prep_q<<<(B_ * P_ * KQ_ + 255) / 256, 256, 0, stream>>>(eln, attr, qin);
  k_pack_e<<<(B_ * NT_ * 4 * 512 + 255) / 256, 256, 0, stream>>>(enc, pe);
  k_pack_w<<<(8 * 4 * 512 + 255) / 256, 256, 0, stream>>>(Wks, E_, 4, pWks);
  k_pack_w<<<(8 * 4 * 512 + 255) / 256, 256, 0, stream>>>(Wvs, E_, 4, pWvs);
  k_pack_w<<<(8 * 4 * 512 + 255) / 256, 256, 0, stream>>>(Wkt, E_, 4, pWkt);
  k_pack_w<<<(8 * 4 * 512 + 255) / 256, 256, 0, stream>>>(Wvt, E_, 4, pWvt);
  k_pack_w<<<(8 * 4 * 512 + 255) / 256, 256, 0, stream>>>(Wcs, E_, 4, pWcs);
  k_pack_w<<<(8 * 4 * 512 + 255) / 256, 256, 0, stream>>>(Wct, E_, 4, pWct);
  k_pack_w<<<(8 * 5 * 512 + 255) / 256, 256, 0, stream>>>(Wqs, E_ + 4, 5, pWqs);
  k_pack_w<<<(8 * 5 * 512 + 255) / 256, 256, 0, stream>>>(Wqt, E_ + 4, 5, pWqt);
  k_pack_w<<<(8 * 5 * 512 + 255) / 256, 256, 0, stream>>>(Wg, E_ + 4, 5, pWg);

  hipMemsetAsync(pks, 0, 4 * PKV, stream);     // zero padding lanes of packed K/V

  k_proj_kv<<<B_ * NT_, 256, 0, stream>>>(enc16, pWks, pWvs, pWkt, pWvt, pks, pvs, pkt, pvt);
  k_proj_q<<<500, 256, 0, stream>>>(qin, pWqs, pWqt, pWg, bg, qs, qt, gate);
  k_attn2<<<504, 256, 0, stream>>>(qs, qt, pks, pvs, pkt, pvt, mask, aso, ato);
  k_combine<<<500, 256, 0, stream>>>(aso, ato, pWcs, bcs, pWct, bct, gate, g16);
  k_pointer<<<NT_, 256, 0, stream>>>(g16, pe, mask, (float*)d_out);
}
